// Multihead_self_attention_7559142441484
// MI455X (gfx1250) — matrix operand order repaired and checked by reading, not compiled, not run on hardware
//
#include <hip/hip_runtime.h>
#include <hip/hip_bf16.h>

// ---------------------------------------------------------------------------
// Multi-head self-attention (b=2, s=2048, d_model=768, h=12, d_k=64) + RoPE
// CDNA5 / gfx1250: bf16 WMMA (v_wmma_f32_16x16x32_bf16) with fp32 accumulate.
// ---------------------------------------------------------------------------

typedef __bf16 bf16_t;
typedef __attribute__((ext_vector_type(16))) __bf16 v16bf;
typedef __attribute__((ext_vector_type(8)))  float  v8f;

#define S_LEN   2048
#define D_MODEL 768
#define NHEADS  12
#define DK      64
#define BATCH   2
#define NTOK    (BATCH * S_LEN)     // 4096
#define LN_THETA_OVER_32 0.28782313662425572f  // ln(10000)/32

union FragB16 { v16bf v; uint4 q[2]; };

// A-fragment (16x32 bf16): lane m = lane&15 picks the row; half = lane>>4
// selects K chunks {half*8..half*8+7} and {16+half*8..16+half*8+7}.
__device__ __forceinline__ v16bf load_fragA(const bf16_t* row, int kbase, int half) {
    FragB16 f;
    f.q[0] = *reinterpret_cast<const uint4*>(row + kbase + half * 8);
    f.q[1] = *reinterpret_cast<const uint4*>(row + kbase + 16 + half * 8);
    return f.v;
}

// B-fragment (32x16 bf16): lane&15 = N column, 16 K values per lane.
__device__ __forceinline__ v16bf load_fragB(const bf16_t* p) {
    FragB16 f;
    f.q[0] = *reinterpret_cast<const uint4*>(p);
    f.q[1] = *reinterpret_cast<const uint4*>(p + 8);
    return f.v;
}

__device__ __forceinline__ v8f wmma_bf16(v16bf a, v16bf b, v8f c) {
    return __builtin_amdgcn_wmma_f32_16x16x32_bf16(false, a, false, b,
                                                   (short)0, c, false, false);
}

// --------------------------- prep kernels ----------------------------------

__global__ void cvt_f32_bf16_kernel(const float* __restrict__ in,
                                    bf16_t* __restrict__ out, int n) {
    int i = blockIdx.x * blockDim.x + threadIdx.x;
    if (i < n) out[i] = (bf16_t)in[i];
}

// out[n*768+k] = in[n*768+k]  (768x768), fp32 -> bf16
__global__ void transpose_w_kernel(const float* __restrict__ in,
                                   bf16_t* __restrict__ out) {
    int i = blockIdx.x * blockDim.x + threadIdx.x;
    if (i < D_MODEL * D_MODEL) {
        int k = i / D_MODEL, n = i % D_MODEL;
        out[i] = (bf16_t)in[k * D_MODEL + n];
    }
}

// --------------------------- QKV + RoPE ------------------------------------
// grid (256 m-tiles, 12 n-groups, 3 which), block 128 (4 waves, one n-tile each)
__global__ void qkv_rope_kernel(const bf16_t* __restrict__ xb,
                                const bf16_t* __restrict__ wqT,
                                const bf16_t* __restrict__ wkT,
                                const bf16_t* __restrict__ wvT,
                                bf16_t* __restrict__ Qb,   // [b,h,s,dk]
                                bf16_t* __restrict__ Kt,   // [b,h,s,dk]
                                bf16_t* __restrict__ Vb) { // [b,h,dk,s]
    const int lane   = threadIdx.x & 31;
    const int wid    = threadIdx.x >> 5;
    const int tile_m = blockIdx.x;               // 0..255
    const int tile_n = blockIdx.y * 4 + wid;     // 0..47
    const int which  = blockIdx.z;               // 0=Q, 1=K, 2=V
    const bf16_t* wT = (which == 0) ? wqT : (which == 1) ? wkT : wvT;

    const int half = lane >> 4;
    const int m    = lane & 15;
    const bf16_t* arow = xb + (size_t)(tile_m * 16 + m) * D_MODEL;
    const int n0 = tile_n * 16;

    v8f acc = {};
    for (int kb = 0; kb < D_MODEL; kb += 32) {
        v16bf a = load_fragA(arow, kb, half);
        v16bf b = load_fragA(wT + (size_t)(n0 + m) * D_MODEL, kb, half);
        acc = wmma_bf16(a, b, acc);
    }

    const int col  = n0 + m;        // C layout: lane&15 == N
    const int head = col >> 6;
    const int fih  = col & 63;      // feature within head

    if (which != 2) {               // RoPE on Q and K
        const int   p2   = fih >> 1;
        const float invf = __expf(-(float)p2 * LN_THETA_OVER_32);
#pragma unroll
        for (int j = 0; j < 8; ++j) {
            int   row   = j + (half << 3);
            int   token = tile_m * 16 + row;
            int   pos   = token & (S_LEN - 1);
            float ang   = (float)pos * invf;
            float cs = __cosf(ang), sn = __sinf(ang);
            float own   = acc[j];
            float other = __shfl_xor(own, 1, 32);   // even<->odd feature pair
            acc[j] = ((lane & 1) == 0) ? (own * cs - other * sn)
                                       : (other * sn + own * cs);
        }
    }

#pragma unroll
    for (int j = 0; j < 8; ++j) {
        int    row   = j + (half << 3);
        int    token = tile_m * 16 + row;
        int    bb    = token >> 11;              // token / S_LEN
        int    pos   = token & (S_LEN - 1);
        int    bh    = bb * NHEADS + head;
        bf16_t val   = (bf16_t)acc[j];
        if (which == 0)      Qb[((size_t)bh * S_LEN + pos) * DK + fih] = val;
        else if (which == 1) Kt[((size_t)bh * S_LEN + pos) * DK + fih] = val;
        else                 Vb[((size_t)bh * DK + fih) * S_LEN + pos] = val;
    }
}

// --------------------------- flash attention -------------------------------
// grid (24 bh, 32 qt-groups), block 128 (4 waves, one 16-query tile each)
__global__ void attn_kernel(const bf16_t* __restrict__ Qb,
                            const bf16_t* __restrict__ Kt,
                            const bf16_t* __restrict__ Vb,
                            bf16_t* __restrict__ Ob) {  // [ntok, d_model]
    __shared__ __align__(16) bf16_t ldsP[4][16][32];

    const int lane = threadIdx.x & 31;
    const int wid  = threadIdx.x >> 5;
    const int bh   = blockIdx.x;                 // 0..23
    const int qt   = blockIdx.y * 4 + wid;       // 0..127
    const int bb   = bh / NHEADS;
    const int hh   = bh % NHEADS;
    const int half = lane >> 4;
    const int m    = lane & 15;

    const bf16_t* Qbase = Qb + (size_t)bh * S_LEN * DK;
    const bf16_t* Kbase = Kt + (size_t)bh * DK * S_LEN;
    const bf16_t* Vbase = Vb + (size_t)bh * S_LEN * DK;

    const int q0 = qt * 16;
    const bf16_t* qrow = Qbase + (size_t)(q0 + m) * DK;
    v16bf aQ0 = load_fragA(qrow, 0, half);
    v16bf aQ1 = load_fragA(qrow, 32, half);

    v8f o[4] = {{}, {}, {}, {}};
    float mstat[8], lstat[8];
#pragma unroll
    for (int j = 0; j < 8; ++j) { mstat[j] = -3.0e38f; lstat[j] = 0.0f; }

    const int numKeys = q0 + 16;                 // causal bound
    const int nblocks = (numKeys + 31) / 32;

    for (int kb = 0; kb < nblocks; ++kb) {
        const int key0 = kb * 32;
        const bool has2 = (key0 + 16) < numKeys; // wave-uniform

        // ---- scores S = Q * K^T for up to 32 keys (two 16-key subtiles) ----
        v8f s0 = {}, s1 = {};
        {
            v16bf b0 = load_fragA(Kbase + (size_t)(key0 + m) * DK, 0, half);
            v16bf b1 = load_fragA(Kbase + (size_t)(key0 + m) * DK, 32, half);
            s0 = wmma_bf16(aQ0, b0, s0);
            s0 = wmma_bf16(aQ1, b1, s0);
        }
        if (has2) {
            v16bf b0 = load_fragA(Kbase + (size_t)(key0 + 16 + m) * DK, 0, half);
            v16bf b1 = load_fragA(Kbase + (size_t)(key0 + 16 + m) * DK, 32, half);
            s1 = wmma_bf16(aQ0, b0, s1);
            s1 = wmma_bf16(aQ1, b1, s1);
        }

        // ---- scale + causal mask (finite -BIG so exp underflows to 0) ----
#pragma unroll
        for (int j = 0; j < 8; ++j) {
            int gq = q0 + j + (half << 3);
            float v0 = s0[j] * 0.125f;                       // 1/sqrt(64)
            if (key0 + m > gq) v0 = -3.0e38f;
            s0[j] = v0;
            float v1 = has2 ? (s1[j] * 0.125f) : -3.0e38f;
            if (key0 + 16 + m > gq) v1 = -3.0e38f;
            s1[j] = v1;
        }

        // ---- online softmax (row lives in one VGPR across a 16-lane half) --
#pragma unroll
        for (int j = 0; j < 8; ++j) {
            float rm = fmaxf(s0[j], s1[j]);
            rm = fmaxf(rm, __shfl_xor(rm, 1, 32));
            rm = fmaxf(rm, __shfl_xor(rm, 2, 32));
            rm = fmaxf(rm, __shfl_xor(rm, 4, 32));
            rm = fmaxf(rm, __shfl_xor(rm, 8, 32));
            float mnew = fmaxf(mstat[j], rm);
            float corr = __expf(mstat[j] - mnew);
            float p0 = __expf(s0[j] - mnew);
            float p1 = __expf(s1[j] - mnew);
            float rs = p0 + p1;
            rs += __shfl_xor(rs, 1, 32);
            rs += __shfl_xor(rs, 2, 32);
            rs += __shfl_xor(rs, 4, 32);
            rs += __shfl_xor(rs, 8, 32);
            lstat[j] = lstat[j] * corr + rs;
            mstat[j] = mnew;
            o[0][j] *= corr; o[1][j] *= corr; o[2][j] *= corr; o[3][j] *= corr;
            // stage P (C layout -> A layout) through LDS
            ldsP[wid][j + (half << 3)][m]      = (bf16_t)p0;
            ldsP[wid][j + (half << 3)][m + 16] = (bf16_t)p1;
        }

        // per-wave LDS store->load handoff (cross-lane): CDNA5 split counter
        asm volatile("s_wait_dscnt 0" ::: "memory");

        v16bf aP = load_fragA(&ldsP[wid][m][0], 0, half);

        // ---- O += P * V  (N tiled over d_k = 4 x 16) ----
#pragma unroll
        for (int t = 0; t < 4; ++t) {
            v16bf bV = load_fragA(Vbase + (size_t)(t * 16 + m) * S_LEN, key0, half);
            o[t] = wmma_bf16(aP, bV, o[t]);
        }
    }

    // ---- normalize and store O in [token, d_model] bf16 for out-proj ----
#pragma unroll
    for (int j = 0; j < 8; ++j) {
        float inv = 1.0f / lstat[j];
        int row   = j + (half << 3);
        int token = bb * S_LEN + q0 + row;
#pragma unroll
        for (int t = 0; t < 4; ++t) {
            int col = hh * DK + t * 16 + m;
            Ob[(size_t)token * D_MODEL + col] = (bf16_t)(o[t][j] * inv);
        }
    }
}

// --------------------------- output projection -----------------------------
// grid (256 m-tiles, 12 n-groups), block 128
__global__ void out_proj_kernel(const bf16_t* __restrict__ Ob,
                                const bf16_t* __restrict__ woT,
                                float* __restrict__ out) {
    const int lane   = threadIdx.x & 31;
    const int wid    = threadIdx.x >> 5;
    const int tile_m = blockIdx.x;
    const int tile_n = blockIdx.y * 4 + wid;
    const int half   = lane >> 4;
    const int m      = lane & 15;
    const bf16_t* arow = Ob + (size_t)(tile_m * 16 + m) * D_MODEL;
    const int n0 = tile_n * 16;

    v8f acc = {};
    for (int kb = 0; kb < D_MODEL; kb += 32) {
        v16bf a = load_fragA(arow, kb, half);
        v16bf b = load_fragA(woT + (size_t)(n0 + m) * D_MODEL, kb, half);
        acc = wmma_bf16(a, b, acc);
    }
#pragma unroll
    for (int j = 0; j < 8; ++j) {
        int row = tile_m * 16 + j + (half << 3);
        out[(size_t)row * D_MODEL + n0 + m] = acc[j];
    }
}

// --------------------------- launch ----------------------------------------

extern "C" void kernel_launch(void* const* d_in, const int* in_sizes, int n_in,
                              void* d_out, int out_size, void* d_ws, size_t ws_size,
                              hipStream_t stream) {
    const float* x  = (const float*)d_in[0];
    const float* wq = (const float*)d_in[1];
    const float* wk = (const float*)d_in[2];
    const float* wv = (const float*)d_in[3];
    const float* wo = (const float*)d_in[4];
    float* out = (float*)d_out;

    char* ws = (char*)d_ws;
    const size_t XN = (size_t)NTOK * D_MODEL;          // 3,145,728 elements
    const size_t WN = (size_t)D_MODEL * D_MODEL;       //   589,824 elements
    bf16_t* xb  = (bf16_t*)ws;                         // 6,291,456 B
    bf16_t* wqT = (bf16_t*)(ws + 2 * XN);
    bf16_t* wkT = wqT + WN;
    bf16_t* wvT = wkT + WN;
    bf16_t* woT = wvT + WN;
    bf16_t* Qb  = woT + WN;                            // [b,h,s,dk]
    bf16_t* Kt  = Qb + XN;                             // [b,h,s,dk]
    bf16_t* Vb  = Kt + XN;                             // [b,h,dk,s]
    bf16_t* Ob  = Vb + XN;                             // [ntok, d_model]
    // total ws use: 2*XN + 4*2*WN + 4*2*XN = ~36.2 MB

    cvt_f32_bf16_kernel<<<(int)((XN + 255) / 256), 256, 0, stream>>>(x, xb, (int)XN);
    int wblocks = (int)((WN + 255) / 256);
    transpose_w_kernel<<<wblocks, 256, 0, stream>>>(wq, wqT);
    transpose_w_kernel<<<wblocks, 256, 0, stream>>>(wk, wkT);
    transpose_w_kernel<<<wblocks, 256, 0, stream>>>(wv, wvT);
    transpose_w_kernel<<<wblocks, 256, 0, stream>>>(wo, woT);

    dim3 g1(NTOK / 16, D_MODEL / (16 * 4), 3);         // (256, 12, 3)
    qkv_rope_kernel<<<g1, 128, 0, stream>>>(xb, wqT, wkT, wvT, Qb, Kt, Vb);

    dim3 g2(BATCH * NHEADS, (S_LEN / 16) / 4);         // (24, 32)
    attn_kernel<<<g2, 128, 0, stream>>>(Qb, Kt, Vb, Ob);

    dim3 g3(NTOK / 16, D_MODEL / (16 * 4));            // (256, 12)
    out_proj_kernel<<<g3, 128, 0, stream>>>(Ob, woT, out);
}
